// ConvVMamba_57449482551785
// MI455X (gfx1250) — matrix operand order repaired and checked by reading, not compiled, not run on hardware
//
#include <hip/hip_runtime.h>
#include <hip/hip_bf16.h>
#include <math.h>

// ---------------------------------------------------------------------------
// CDNA5 (gfx1250) implementation of the ConvVMamba forward pass.
// All 1x1 convs -> WMMA f16 GEMMs (f32 accum) with TDM-staged LDS tiles.
// ---------------------------------------------------------------------------

typedef __attribute__((ext_vector_type(16))) _Float16   v16h;
typedef __attribute__((ext_vector_type(8)))  float      v8f;
typedef __attribute__((ext_vector_type(4)))  unsigned   v4u;
typedef __attribute__((ext_vector_type(8)))  int        v8i;
typedef __attribute__((ext_vector_type(4)))  int        v4i;

#ifndef HAVE_TDM
#  if defined(__has_builtin)
#    if __has_builtin(__builtin_amdgcn_tensor_load_to_lds)
#      define HAVE_TDM 1
#    endif
#  endif
#endif
#ifndef HAVE_TDM
#  define HAVE_TDM 0
#endif

#define ACT_NONE     0
#define ACT_GELU     1
#define ACT_SILU     2
#define ACT_SOFTPLUS 3

#define TILE_N   256            // pixels per block (2 x 16 per wave, 8 waves)
#define TILE_K   32
#define TILE_BYTES (TILE_K * TILE_N * 4)

__device__ __forceinline__ float gelu_f(float x) {
  return 0.5f * x * (1.0f + erff(x * 0.70710678118654752f));
}
__device__ __forceinline__ float silu_f(float x) {
  return x / (1.0f + __expf(-x));
}
__device__ __forceinline__ float apply_act(float v, int act) {
  if (act == ACT_GELU)     return gelu_f(v);
  if (act == ACT_SILU)     return silu_f(v);
  if (act == ACT_SOFTPLUS) return (v > 20.0f) ? v : log1pf(__expf(v));
  return v;
}

// ---------------------------------------------------------------------------
// TDM: load a [32 rows x 256 cols] f32 2D tile (row stride = L elements) into
// LDS at byte offset lds_off.  Descriptor packing per CDNA5 ISA.
// Passing the real (ptrtoint-derived) LDS offset also captures the __shared__
// array into the side-effecting intrinsic, so the compiler keeps the ds_loads.
// ---------------------------------------------------------------------------
#if HAVE_TDM
__device__ __forceinline__ void tdm_load_tile(unsigned lds_off, const float* src,
                                              long rem_cols, int rem_rows,
                                              int row_stride_elems) {
  if (rem_rows < 0) rem_rows = 0;
  unsigned long long ga = (unsigned long long)(__SIZE_TYPE__)src;
  unsigned td0 = (rem_cols > 0x7FFFFFFFL) ? 0x7FFFFFFFu : (unsigned)rem_cols;
  unsigned td1 = (unsigned)rem_rows;
  unsigned long long st = (unsigned long long)(unsigned)row_stride_elems;
  v4u g0;
  g0.x = 1u;                                            // count=1, user mode
  g0.y = lds_off;                                       // lds_addr (bytes)
  g0.z = (unsigned)ga;                                  // global_addr[31:0]
  g0.w = (unsigned)((ga >> 32) & 0x01FFFFFFu) | 0x80000000u;  // addr[56:32], type=2
  v8i g1;
  g1[0] = (int)(2u << 16);                              // data_size = 4B
  g1[1] = (int)((td0 & 0xFFFFu) << 16);                 // tensor_dim0[15:0]
  g1[2] = (int)(((td0 >> 16) & 0xFFFFu) | ((td1 & 0xFFFFu) << 16));
  g1[3] = (int)(((td1 >> 16) & 0xFFFFu) | ((unsigned)TILE_N << 16)); // tile_dim0
  g1[4] = (int)(unsigned)TILE_K;                        // tile_dim1
  g1[5] = (int)(unsigned)(st & 0xFFFFFFFFu);            // dim0 stride lo
  g1[6] = (int)(unsigned)((st >> 32) & 0xFFFFu);        // dim0 stride hi
  g1[7] = 0;
  v4i gz = {0, 0, 0, 0};
#if defined(__clang_major__) && __clang_major__ >= 23
  v8i gz8 = {0, 0, 0, 0, 0, 0, 0, 0};
  __builtin_amdgcn_tensor_load_to_lds(g0, g1, gz, gz, gz8, 0);
#else
  __builtin_amdgcn_tensor_load_to_lds(g0, g1, gz, gz, 0);
#endif
}
#endif

// ---------------------------------------------------------------------------
// Pointwise (1x1) conv as WMMA GEMM:  Y[n,co,l] = sum_ci W[co,ci]*X[n,ci,l]
// + bias, activation, optional residual.  Block = 256 thr = 8 waves.
// Each wave computes two adjacent 16x16 tiles (reusing the A fragment);
// the block covers 256 pixels.  grid = (L/256, ceil(Cout/16), Nbatch)
// ---------------------------------------------------------------------------
__global__ __launch_bounds__(256) void gemm_pw_wmma_kernel(
    const float* __restrict__ X, const float* __restrict__ Wm,
    const float* __restrict__ bias, const float* __restrict__ Res,
    float* __restrict__ Y, int Cin, int Cout, int L, int act) {
  __shared__ float lds[2][TILE_K * TILE_N];
  const int lane = threadIdx.x & 31;
  const int wid  = threadIdx.x >> 5;
  const long n0b = (long)blockIdx.x * TILE_N;
  const int  co0 = blockIdx.y * 16;
  const float* Xb = X + (size_t)blockIdx.z * Cin * (size_t)L;
  const size_t bofY = (size_t)blockIdx.z * Cout * (size_t)L;
  const int ksteps = (Cin + 31) >> 5;
  const long Lrem = (long)L - n0b;
  // Low 32 bits of the generic address of a __shared__ object == LDS byte
  // offset (ISA: LDS aperture maps addr[31:0] -> LDS offset).
  const unsigned lds_base = (unsigned)(__SIZE_TYPE__)(void*)&lds[0][0];

#if HAVE_TDM
  if (wid == 0) tdm_load_tile(lds_base, Xb + n0b, Lrem, Cin, L);
#endif

  // A fragment geometry (16-bit 16x32 layout): lanes 0-15 K-base 0, lanes
  // 16-31 K-base 8; halves 0..7 -> K=base+h, halves 8..15 -> K=base+h+8.
  const int mrow  = lane & 15;
  const int abase = (lane >> 4) * 8;
  const int arow  = co0 + mrow;
  const bool mok  = arow < Cout;
  const float* wrow = Wm + (size_t)(mok ? arow : 0) * Cin;

  v8f c0 = {}, c1 = {};
  for (int s = 0; s < ksteps; ++s) {
    const int k0 = s << 5;
    const int buf = s & 1;
#if HAVE_TDM
    if (wid == 0) {
      if (s + 1 < ksteps) {
        tdm_load_tile(lds_base + (unsigned)(((s + 1) & 1) * TILE_BYTES),
                      Xb + (size_t)(k0 + 32) * L + n0b, Lrem, Cin - (k0 + 32), L);
        __builtin_amdgcn_s_wait_tensorcnt(1);
      } else {
        __builtin_amdgcn_s_wait_tensorcnt(0);
      }
    }
    __syncthreads();
#else
    __syncthreads();
    for (int i = threadIdx.x; i < TILE_K * TILE_N; i += 256) {
      int r = i / TILE_N, cc = i % TILE_N;
      lds[buf][i] = (k0 + r < Cin) ? Xb[(size_t)(k0 + r) * L + n0b + cc] : 0.0f;
    }
    __syncthreads();
#endif
    // A fragment: unconditional clamped loads, then zero-select (keeps the 16
    // global loads batchable -- no exec-masked per-element loads/waits).
    v16h a, b0, b1;
#pragma unroll
    for (int h = 0; h < 16; ++h) {
      int kk = k0 + abase + h + ((h & 8) ? 8 : 0);
      float av = wrow[kk < Cin ? kk : 0];
      if (!mok || kk >= Cin) av = 0.0f;
      a[h] = (_Float16)av;
    }
    // B fragments: 32x16 layout; lanes 0-15 hold K 0..7,16..23, lanes 16-31 hold
    // K 8..15,24..31; column = lane&15.  Two tiles at col and col+16.
    const float* bsrc =
        &lds[buf][((lane >> 4) * 8) * TILE_N + wid * 32 + (lane & 15)];
#pragma unroll
    for (int h = 0; h < 16; ++h) {
      b0[h] = (_Float16)bsrc[(h + ((h & 8) ? 8 : 0)) * TILE_N];
      b1[h] = (_Float16)bsrc[(h + ((h & 8) ? 8 : 0)) * TILE_N + 16];
    }
    c0 = __builtin_amdgcn_wmma_f32_16x16x32_f16(false, a, false, b0, (short)0, c0,
                                                false, false);
    c1 = __builtin_amdgcn_wmma_f32_16x16x32_f16(false, a, false, b1, (short)0, c1,
                                                false, false);
    __syncthreads();
  }

  // D store: element (m,n): m = r + 8*(lane>>4), n = lane&15
  const long col = n0b + wid * 32 + (lane & 15);
  const int mb = (lane >> 4) * 8;
#pragma unroll
  for (int r = 0; r < 8; ++r) {
    int co = co0 + mb + r;
    if (co < Cout) {
      size_t idx0 = bofY + (size_t)co * L + col;
      float v0 = c0[r], v1 = c1[r];
      if (bias) { float bb = bias[co]; v0 += bb; v1 += bb; }
      v0 = apply_act(v0, act);
      v1 = apply_act(v1, act);
      if (Res) { v0 += Res[idx0]; v1 += Res[idx0 + 16]; }
      Y[idx0] = v0;
      Y[idx0 + 16] = v1;
    }
  }
}

// ---------------------------------------------------------------------------
// Channel LayerNorm at each pixel (axis=1 of NCHW).
// ---------------------------------------------------------------------------
__global__ void ln2d_kernel(const float* __restrict__ X, const float* __restrict__ g,
                            const float* __restrict__ bt, float* __restrict__ Y,
                            int C, long HW, long total) {
  long idx = (long)blockIdx.x * blockDim.x + threadIdx.x;
  if (idx >= total) return;
  long n = idx / HW, l = idx % HW;
  const float* xb = X + (size_t)n * C * HW + l;
  float s = 0.f, ss = 0.f;
  for (int cidx = 0; cidx < C; ++cidx) {
    float v = xb[(size_t)cidx * HW];
    s += v; ss += v * v;
  }
  float m = s / C;
  float var = ss / C - m * m;
  if (var < 0.f) var = 0.f;
  float inv = rsqrtf(var + 1e-5f);
  float* yb = Y + (size_t)n * C * HW + l;
  for (int cidx = 0; cidx < C; ++cidx) {
    float v = xb[(size_t)cidx * HW];
    yb[(size_t)cidx * HW] = (v - m) * inv * g[cidx] + bt[cidx];
  }
}

// ---------------------------------------------------------------------------
// Depthwise KxK conv, optional fused BN affine and activation.
// ---------------------------------------------------------------------------
__global__ void dwconv_kernel(const float* __restrict__ X, const float* __restrict__ Wd,
                              const float* __restrict__ Bd,
                              const float* __restrict__ bn_m, const float* __restrict__ bn_v,
                              const float* __restrict__ bn_w, const float* __restrict__ bn_b,
                              float* __restrict__ Y, int C, int H, int W, int K, int P,
                              int act, long total) {
  long idx = (long)blockIdx.x * blockDim.x + threadIdx.x;
  if (idx >= total) return;
  int w = (int)(idx % W); long t = idx / W;
  int h = (int)(t % H); t /= H;
  int cc = (int)(t % C); long n = t / C;
  const float* xb = X + ((size_t)n * C + cc) * H * (size_t)W;
  float s = Bd ? Bd[cc] : 0.0f;
  for (int kh = 0; kh < K; ++kh) {
    int ih = h - P + kh;
    if (ih < 0 || ih >= H) continue;
    for (int kw = 0; kw < K; ++kw) {
      int iw = w - P + kw;
      if (iw < 0 || iw >= W) continue;
      s += xb[(size_t)ih * W + iw] * Wd[(cc * K + kh) * K + kw];
    }
  }
  if (bn_v) {
    float sc = bn_w[cc] * rsqrtf(bn_v[cc] + 1e-5f);
    s = (s - bn_m[cc]) * sc + bn_b[cc];
  }
  Y[idx] = apply_act(s, act);
}

// ---------------------------------------------------------------------------
// Dense 3x3 stride-2 conv (stem & downsamples), padding=1.
// ---------------------------------------------------------------------------
__global__ void conv3s2_kernel(const float* __restrict__ X, const float* __restrict__ Wc,
                               const float* __restrict__ Bc, float* __restrict__ Y,
                               int Cin, int Cout, int Hin, int Win, long total) {
  long idx = (long)blockIdx.x * blockDim.x + threadIdx.x;
  if (idx >= total) return;
  int Hout = Hin >> 1, Wout = Win >> 1;
  int ow = (int)(idx % Wout); long t = idx / Wout;
  int oh = (int)(t % Hout); t /= Hout;
  int co = (int)(t % Cout); long n = t / Cout;
  float s = Bc ? Bc[co] : 0.0f;
  const float* xb = X + (size_t)n * Cin * Hin * (size_t)Win;
  for (int ci = 0; ci < Cin; ++ci) {
    const float* xc = xb + (size_t)ci * Hin * Win;
    const float* wc = Wc + ((size_t)co * Cin + ci) * 9;
    for (int kh = 0; kh < 3; ++kh) {
      int ih = oh * 2 - 1 + kh;
      if (ih < 0 || ih >= Hin) continue;
      for (int kw = 0; kw < 3; ++kw) {
        int iw = ow * 2 - 1 + kw;
        if (iw < 0 || iw >= Win) continue;
        s += xc[(size_t)ih * Win + iw] * wc[kh * 3 + kw];
      }
    }
  }
  Y[idx] = s;
}

// 2x bilinear downsample (exact 2x2 average) for positional embedding.
__global__ void avgpool2_kernel(const float* __restrict__ X, float* __restrict__ Y,
                                int H, int W, long total) {
  long idx = (long)blockIdx.x * blockDim.x + threadIdx.x;
  if (idx >= total) return;
  int Wo = W >> 1;
  int ow = (int)(idx % Wo); long t = idx / Wo;
  int oh = (int)(t % (H >> 1)); long cc = t / (H >> 1);
  const float* xb = X + (size_t)cc * H * W + (size_t)(2 * oh) * W + 2 * ow;
  Y[idx] = 0.25f * (xb[0] + xb[1] + xb[W] + xb[W + 1]);
}

__global__ void addpe_kernel(float* __restrict__ X, const float* __restrict__ pe,
                             const float* __restrict__ gamma, int C, long HW, long total) {
  long idx = (long)blockIdx.x * blockDim.x + threadIdx.x;
  if (idx >= total) return;
  long l = idx % HW;
  int cc = (int)((idx / HW) % C);
  X[idx] += pe[(size_t)cc * HW + l] * gamma[cc];
}

// 2x bilinear upsample (half-pixel, edge clamp), accumulate into Y.
__global__ void up2add_kernel(float* __restrict__ Y, const float* __restrict__ X,
                              int H, int W, long total) {
  long idx = (long)blockIdx.x * blockDim.x + threadIdx.x;
  if (idx >= total) return;
  int W2 = 2 * W, H2 = 2 * H;
  int ow = (int)(idx % W2); long t = idx / W2;
  int oh = (int)(t % H2); long nc = t / H2;
  float sx = ow * 0.5f - 0.25f, sy = oh * 0.5f - 0.25f;
  int ix0 = (int)floorf(sx), iy0 = (int)floorf(sy);
  float fx = sx - ix0, fy = sy - iy0;
  int ix1 = ix0 + 1, iy1 = iy0 + 1;
  ix0 = ix0 < 0 ? 0 : (ix0 > W - 1 ? W - 1 : ix0);
  ix1 = ix1 < 0 ? 0 : (ix1 > W - 1 ? W - 1 : ix1);
  iy0 = iy0 < 0 ? 0 : (iy0 > H - 1 ? H - 1 : iy0);
  iy1 = iy1 < 0 ? 0 : (iy1 > H - 1 ? H - 1 : iy1);
  const float* xb = X + (size_t)nc * H * W;
  float v00 = xb[(size_t)iy0 * W + ix0], v01 = xb[(size_t)iy0 * W + ix1];
  float v10 = xb[(size_t)iy1 * W + ix0], v11 = xb[(size_t)iy1 * W + ix1];
  float v = (1 - fy) * ((1 - fx) * v00 + fx * v01) + fy * ((1 - fx) * v10 + fx * v11);
  Y[idx] += v;
}

// xs: four scan orderings of h -> xs[n,k,d,l]
__global__ void make_xs_kernel(const float* __restrict__ h, float* __restrict__ xs,
                               int D, int H, int W, long total) {
  long idx = (long)blockIdx.x * blockDim.x + threadIdx.x;
  if (idx >= total) return;
  long L = (long)H * W;
  long l = idx % L; long t = idx / L;
  int d = (int)(t % D); long n = t / D;
  const float* hp = h + ((size_t)n * D + d) * L;
  float v0 = hp[l];
  int w = (int)(l / H), hh = (int)(l % H);          // l interpreted as l' for x1
  float v1 = hp[(size_t)hh * W + w];
  float* base = xs + ((size_t)n * 4 * D + d) * L;
  size_t kstride = (size_t)D * L;
  base[l] = v0;
  base[kstride + l] = v1;
  base[2 * kstride + (L - 1 - l)] = v0;
  base[3 * kstride + (L - 1 - l)] = v1;
}

// Sequential selective scan (N=1 state) per (n,k,d).
__global__ void ss_scan_kernel(const float* __restrict__ delta, const float* __restrict__ xs,
                               const float* __restrict__ xdbl,
                               const float* __restrict__ A_logs, const float* __restrict__ Ds,
                               float* __restrict__ y, int D, int Rt, int L) {
  int idx = blockIdx.x * blockDim.x + threadIdx.x;
  if (idx >= 16 * D) return;
  int d = idx % D;
  int k = (idx / D) & 3;
  int n = idx / (4 * D);
  size_t chan = ((size_t)(n * 4 + k) * D + d) * (size_t)L;
  const float* dl = delta + chan;
  const float* u  = xs + chan;
  const float* Bs = xdbl + ((size_t)(n * 4 + k) * Rt + (Rt - 2)) * (size_t)L;
  const float* Cs = Bs + L;
  float A  = -__expf(A_logs[k * D + d]);
  float Dv = Ds[k * D + d];
  float* yo = y + chan;
  float hst = 0.0f;
  for (int l = 0; l < L; ++l) {
    float dlt = dl[l], uu = u[l];
    float dA = __expf(dlt * A);
    hst = dA * hst + dlt * Bs[l] * uu;
    yo[l] = hst * Cs[l] + Dv * uu;
  }
}

// ym[l=(h,w)] = y0[l] + y2[L-1-l] + y1[w*H+h] + y3[L-1-(w*H+h)]
__global__ void merge_y_kernel(const float* __restrict__ y, float* __restrict__ ym,
                               int D, int H, int W, long total) {
  long idx = (long)blockIdx.x * blockDim.x + threadIdx.x;
  if (idx >= total) return;
  long L = (long)H * W;
  long l = idx % L; long t = idx / L;
  int d = (int)(t % D); long n = t / D;
  int hh = (int)(l / W), w = (int)(l % W);
  long lp = (long)w * H + hh;
  const float* yb = y + ((size_t)n * 4 * D + d) * L;
  size_t ks = (size_t)D * L;
  float v = yb[l] + yb[2 * ks + (L - 1 - l)] + yb[ks + lp] + yb[3 * ks + (L - 1 - lp)];
  ym[((size_t)n * D + d) * L + l] = v;
}

// ---------------------------------------------------------------------------
// Host orchestration
// ---------------------------------------------------------------------------
#define ELT(kern, total, ...)                                             \
  do {                                                                    \
    long _t = (long)(total);                                              \
    int _b = (int)((_t + 255) / 256);                                     \
    kern<<<_b, 256, 0, stream>>>(__VA_ARGS__);                            \
  } while (0)

static inline void gemm_pw(const float* X, const float* Wm, const float* bias,
                           const float* Res, float* Y, int Cin, int Cout, int L,
                           int Nb, int act, hipStream_t stream) {
  dim3 g(L / TILE_N, (Cout + 15) / 16, Nb);
  gemm_pw_wmma_kernel<<<g, 256, 0, stream>>>(X, Wm, bias, Res, Y, Cin, Cout, L, act);
}

// flattened input indices (jax pytree: dict keys sorted, lists in order)
enum {
  I_X0 = 0, I_X1,
  I_LAT1_B, I_LAT1_W, I_LAT2_B, I_LAT2_W,
  I_O1_BN_B, I_O1_BN_M, I_O1_BN_V, I_O1_BN_W, I_O1_DW_B, I_O1_DW_W, I_O1_PW_B, I_O1_PW_W,
  I_O2_BN_B, I_O2_BN_M, I_O2_BN_V, I_O2_BN_W, I_O2_DW_B, I_O2_DW_W, I_O2_PW_B, I_O2_PW_W,
  I_OUT3_B, I_OUT3_W, I_PE_EMB, I_PE_GAMMA,
  I_S0_DW_B, I_S0_DW_W, I_S0_FC1_B, I_S0_FC1_W, I_S0_FC2_B, I_S0_FC2_W, I_S0_LN_B, I_S0_LN_W,
  I_STG0 = 34,  // each stage: 33 entries (see offsets below)
  I_STG1 = 67,
  I_STEM_LN_B = 100, I_STEM_LN_W = 101, I_STEM_W = 102
};
// stage-relative offsets
enum {
  SCV_DW_B = 0, SCV_DW_W, SCV_FC1_B, SCV_FC1_W, SCV_FC2_B, SCV_FC2_W, SCV_LN_B, SCV_LN_W,
  SDS_B, SDS_LN_B, SDS_LN_W, SDS_W,
  SV_FC1_B, SV_FC1_W, SV_FC2_B, SV_FC2_W, SV_LN1_B, SV_LN1_W, SV_LN2_B, SV_LN2_W,
  SS_ALOG, SS_DS, SS_DT_B, SS_DT_W, SS_DW_B, SS_DW_W, SS_IN_B, SS_IN_W,
  SS_OLN_B, SS_OLN_W, SS_OUT_B, SS_OUT_W, SS_XPROJ
};

extern "C" void kernel_launch(void* const* d_in, const int* in_sizes, int n_in,
                              void* d_out, int out_size, void* d_ws, size_t ws_size,
                              hipStream_t stream) {
  (void)in_sizes; (void)n_in; (void)out_size; (void)ws_size;
  auto P = [&](int i) -> const float* { return (const float*)d_in[i]; };

  const int N = 4;
  const int C0 = 48, H0 = 256, W0 = 256; const long L0 = (long)H0 * W0;
  const int C1 = 96, H1 = 128, W1 = 128; const long L1 = (long)H1 * W1;
  const int C2 = 192, H2 = 64, W2 = 64;  const long L2 = (long)H2 * W2;

  float* ws = (float*)d_ws;
  size_t cur = 0;
  auto alloc = [&](size_t nelts) -> float* {
    float* p = ws + cur;
    cur += (nelts + 63) & ~(size_t)63;
    return p;
  };

  // persistent feature maps
  float* f0 = alloc((size_t)N * C0 * L0);
  float* f1 = alloc((size_t)N * C1 * L1);
  float* f2 = alloc((size_t)N * C2 * L2);
  const size_t mark = cur;

  // ---- stem -------------------------------------------------------------
  {
    float* t_in = alloc((size_t)N * 512 * 512);
    size_t half = (size_t)2 * 512 * 512;
    hipMemcpyAsync(t_in, P(I_X0), half * sizeof(float), hipMemcpyDeviceToDevice, stream);
    hipMemcpyAsync(t_in + half, P(I_X1), half * sizeof(float), hipMemcpyDeviceToDevice, stream);
    ELT(conv3s2_kernel, (long)N * C0 * L0, t_in, P(I_STEM_W), nullptr, f0, 1, C0, 512, 512,
        (long)N * C0 * L0);
    ELT(ln2d_kernel, (long)N * L0, f0, P(I_STEM_LN_W), P(I_STEM_LN_B), f0, C0, L0, (long)N * L0);

    // stage0 ConvNeXt block
    float* t1 = alloc((size_t)N * C0 * L0);
    float* t4 = alloc((size_t)N * 4 * C0 * L0);
    ELT(dwconv_kernel, (long)N * C0 * L0, f0, P(I_S0_DW_W), P(I_S0_DW_B),
        nullptr, nullptr, nullptr, nullptr, t1, C0, H0, W0, 7, 3, ACT_NONE, (long)N * C0 * L0);
    ELT(ln2d_kernel, (long)N * L0, t1, P(I_S0_LN_W), P(I_S0_LN_B), t1, C0, L0, (long)N * L0);
    gemm_pw(t1, P(I_S0_FC1_W), P(I_S0_FC1_B), nullptr, t4, C0, 4 * C0, (int)L0, N, ACT_GELU, stream);
    gemm_pw(t4, P(I_S0_FC2_W), P(I_S0_FC2_B), f0, f0, 4 * C0, C0, (int)L0, N, ACT_NONE, stream);

    // positional embedding: 2x down of (48,512,512) then x += pe*gamma
    float* pe_h = alloc((size_t)C0 * L0);
    ELT(avgpool2_kernel, (long)C0 * L0, P(I_PE_EMB), pe_h, 512, 512, (long)C0 * L0);
    ELT(addpe_kernel, (long)N * C0 * L0, f0, pe_h, P(I_PE_GAMMA), C0, L0, (long)N * C0 * L0);
  }
  cur = mark;

  // ---- two stages: downsample + ConvNeXt + VSS ---------------------------
  for (int st = 0; st < 2; ++st) {
    const int base = (st == 0) ? I_STG0 : I_STG1;
    const int ci = (st == 0) ? C0 : C1;
    const int co = (st == 0) ? C1 : C2;
    const int Hin = (st == 0) ? H0 : H1, Win = Hin;
    const int Ho = Hin >> 1, Wo = Ho;
    const long Lin = (long)Hin * Win, Lo = (long)Ho * Wo;
    float* fi = (st == 0) ? f0 : f1;
    float* fo = (st == 0) ? f1 : f2;
    auto SP = [&](int off) { return P(base + off); };

    // downsample: LN then conv3x3 s2
    float* t_ln = alloc((size_t)N * ci * Lin);
    ELT(ln2d_kernel, (long)N * Lin, fi, SP(SDS_LN_W), SP(SDS_LN_B), t_ln, ci, Lin, (long)N * Lin);
    ELT(conv3s2_kernel, (long)N * co * Lo, t_ln, SP(SDS_W), SP(SDS_B), fo, ci, co, Hin, Win,
        (long)N * co * Lo);

    // ConvNeXt block on fo
    float* t1 = alloc((size_t)N * co * Lo);
    float* t4 = alloc((size_t)N * 4 * co * Lo);
    ELT(dwconv_kernel, (long)N * co * Lo, fo, SP(SCV_DW_W), SP(SCV_DW_B),
        nullptr, nullptr, nullptr, nullptr, t1, co, Ho, Wo, 7, 3, ACT_NONE, (long)N * co * Lo);
    ELT(ln2d_kernel, (long)N * Lo, t1, SP(SCV_LN_W), SP(SCV_LN_B), t1, co, Lo, (long)N * Lo);
    gemm_pw(t1, SP(SCV_FC1_W), SP(SCV_FC1_B), nullptr, t4, co, 4 * co, (int)Lo, N, ACT_GELU, stream);
    gemm_pw(t4, SP(SCV_FC2_W), SP(SCV_FC2_B), fo, fo, 4 * co, co, (int)Lo, N, ACT_NONE, stream);

    // ---- VSS block: x = x + ss2d(ln1(x)) -------------------------------
    const int D = co;
    const int R = (D + 15) / 16;
    const int Rt = R + 2;
    float* t_ln1 = t1;  // reuse
    ELT(ln2d_kernel, (long)N * Lo, fo, SP(SV_LN1_W), SP(SV_LN1_B), t_ln1, co, Lo, (long)N * Lo);

    float* t_h  = alloc((size_t)N * D * Lo);
    gemm_pw(t_ln1, SP(SS_IN_W), SP(SS_IN_B), nullptr, t_h, D, D, (int)Lo, N, ACT_NONE, stream);
    float* t_hs = alloc((size_t)N * D * Lo);
    ELT(dwconv_kernel, (long)N * D * Lo, t_h, SP(SS_DW_W), SP(SS_DW_B),
        nullptr, nullptr, nullptr, nullptr, t_hs, D, Ho, Wo, 3, 1, ACT_SILU, (long)N * D * Lo);

    float* t_xs = alloc((size_t)N * 4 * D * Lo);
    ELT(make_xs_kernel, (long)N * D * Lo, t_hs, t_xs, D, Ho, Wo, (long)N * D * Lo);

    float* t_xd = alloc((size_t)N * 4 * Rt * Lo);
    for (int n = 0; n < N; ++n)
      for (int k = 0; k < 4; ++k)
        gemm_pw(t_xs + ((size_t)(n * 4 + k) * D) * Lo, SP(SS_XPROJ) + (size_t)k * Rt * D,
                nullptr, nullptr, t_xd + ((size_t)(n * 4 + k) * Rt) * Lo,
                D, Rt, (int)Lo, 1, ACT_NONE, stream);

    float* t_dl = alloc((size_t)N * 4 * D * Lo);
    for (int n = 0; n < N; ++n)
      for (int k = 0; k < 4; ++k)
        gemm_pw(t_xd + ((size_t)(n * 4 + k) * Rt) * Lo, SP(SS_DT_W) + (size_t)k * D * R,
                SP(SS_DT_B) + (size_t)k * D, nullptr,
                t_dl + ((size_t)(n * 4 + k) * D) * Lo, R, D, (int)Lo, 1, ACT_SOFTPLUS, stream);

    float* t_y = alloc((size_t)N * 4 * D * Lo);
    {
      int nth = 16 * D;
      ss_scan_kernel<<<(nth + 127) / 128, 128, 0, stream>>>(
          t_dl, t_xs, t_xd, SP(SS_ALOG), SP(SS_DS), t_y, D, Rt, (int)Lo);
    }

    float* t_ym = alloc((size_t)N * D * Lo);
    ELT(merge_y_kernel, (long)N * D * Lo, t_y, t_ym, D, Ho, Wo, (long)N * D * Lo);
    ELT(ln2d_kernel, (long)N * Lo, t_ym, SP(SS_OLN_W), SP(SS_OLN_B), t_ym, D, Lo, (long)N * Lo);
    gemm_pw(t_ym, SP(SS_OUT_W), SP(SS_OUT_B), fo, fo, D, D, (int)Lo, N, ACT_NONE, stream);

    // VSS MLP: x = x + fc2(gelu(fc1(ln2(x))))
    ELT(ln2d_kernel, (long)N * Lo, fo, SP(SV_LN2_W), SP(SV_LN2_B), t1, co, Lo, (long)N * Lo);
    gemm_pw(t1, SP(SV_FC1_W), SP(SV_FC1_B), nullptr, t4, co, 4 * co, (int)Lo, N, ACT_GELU, stream);
    gemm_pw(t4, SP(SV_FC2_W), SP(SV_FC2_B), fo, fo, 4 * co, co, (int)Lo, N, ACT_NONE, stream);

    cur = mark;
  }

  // ---- FPN decode --------------------------------------------------------
  {
    float* t_o2 = alloc((size_t)N * C2 * L2);
    gemm_pw(f2, P(I_OUT3_W), P(I_OUT3_B), nullptr, t_o2, C2, C2, (int)L2, N, ACT_NONE, stream);

    float* t_m1 = alloc((size_t)N * C2 * L1);
    gemm_pw(f1, P(I_LAT2_W), P(I_LAT2_B), nullptr, t_m1, C1, C2, (int)L1, N, ACT_NONE, stream);
    ELT(up2add_kernel, (long)N * C2 * L1, t_m1, t_o2, H2, W2, (long)N * C2 * L1);

    // head out2: (192 -> 96) on m1
    float* t_h1 = alloc((size_t)N * C2 * L1);
    ELT(dwconv_kernel, (long)N * C2 * L1, t_m1, P(I_O2_DW_W), P(I_O2_DW_B),
        P(I_O2_BN_M), P(I_O2_BN_V), P(I_O2_BN_W), P(I_O2_BN_B),
        t_h1, C2, H1, W1, 3, 1, ACT_GELU, (long)N * C2 * L1);
    float* t_o1 = alloc((size_t)N * C1 * L1);
    gemm_pw(t_h1, P(I_O2_PW_W), P(I_O2_PW_B), nullptr, t_o1, C2, C1, (int)L1, N, ACT_NONE, stream);

    float* t_m0 = alloc((size_t)N * C1 * L0);
    gemm_pw(f0, P(I_LAT1_W), P(I_LAT1_B), nullptr, t_m0, C0, C1, (int)L0, N, ACT_NONE, stream);
    ELT(up2add_kernel, (long)N * C1 * L0, t_m0, t_o1, H1, W1, (long)N * C1 * L0);

    // head out1: (96 -> 48) on m0 -> final output
    float* t_h0 = alloc((size_t)N * C1 * L0);
    ELT(dwconv_kernel, (long)N * C1 * L0, t_m0, P(I_O1_DW_W), P(I_O1_DW_B),
        P(I_O1_BN_M), P(I_O1_BN_V), P(I_O1_BN_W), P(I_O1_BN_B),
        t_h0, C1, H0, W0, 3, 1, ACT_GELU, (long)N * C1 * L0);
    gemm_pw(t_h0, P(I_O1_PW_W), P(I_O1_PW_B), nullptr, (float*)d_out, C1, C0, (int)L0, N,
            ACT_NONE, stream);
  }
}
